// MultiHeadVer2_89395449299512
// MI455X (gfx1250) — matrix operand order repaired and checked by reading, not compiled, not run on hardware
//
#include <hip/hip_runtime.h>
#include <hip/hip_bf16.h>

typedef __attribute__((ext_vector_type(16))) _Float16 v16h;
typedef __attribute__((ext_vector_type(8)))  _Float16 v8h;
typedef __attribute__((ext_vector_type(8)))  float    v8f;
typedef __attribute__((ext_vector_type(4)))  int      v4i;

#define EMBED 1024
#define HEADS 16
#define HSIZE 64
#define BATCH 4
#define SEQ   2048

// ---------------------------------------------------------------------------
// gfx1250 async global->LDS copy (ASYNCcnt-tracked), with sync fallback.
// Builtin signature (probe-confirmed): (v4i AS1* gsrc, v4i AS3* ldst, Ii, Ii)
// ---------------------------------------------------------------------------
#if defined(__has_builtin)
#  if __has_builtin(__builtin_amdgcn_global_load_async_to_lds_b128)
#    define HAVE_ASYNC 1
#  endif
#endif
#ifndef HAVE_ASYNC
#  define HAVE_ASYNC 0
#endif

#if HAVE_ASYNC
#  define ACP(l, g)                                                        \
    __builtin_amdgcn_global_load_async_to_lds_b128(                        \
        (__attribute__((address_space(1))) v4i*)(g),                       \
        (__attribute__((address_space(3))) v4i*)(l), 0, 0)
#  if __has_builtin(__builtin_amdgcn_s_wait_asynccnt)
#    define WAIT_ASYNC(n) __builtin_amdgcn_s_wait_asynccnt(n)
#  else
#    define WAIT_ASYNC(n) asm volatile("s_wait_asynccnt %0" ::"i"(n) : "memory")
#  endif
#else
#  define ACP(l, g) (*(v8h*)(l) = *(const v8h*)(g))   // 16B sync copy
#  define WAIT_ASYNC(n) ((void)0)
#endif

// ---------------------------------------------------------------------------
// A-fragment for v_wmma_f32_16x16x32_f16 (16x32 MxK), in this file's contiguous k order (not the ISA table order):
// lanes 0-15 -> M=lane, K {0..15}; lanes 16-31 -> M=lane-16, K {16..31}
// => two contiguous 16B loads at kA, kA+8 halves; kA = (lane<16)?0:16.
// ---------------------------------------------------------------------------
static __device__ __forceinline__ v16h make_a_frag(const _Float16* rowk0, int kA) {
  v8h lo = *(const v8h*)(rowk0 + kA);
  v8h hi = *(const v8h*)(rowk0 + kA + 8);
  v16h a;
#pragma unroll
  for (int i = 0; i < 8; ++i) { a[i] = lo[i]; a[i + 8] = hi[i]; }
  return a;
}

// 16 contiguous halves as two 16B loads (keeps 16B alignment on padded LDS rows)
static __device__ __forceinline__ v16h ld_b16(const _Float16* p) {
  v8h lo = *(const v8h*)p;
  v8h hi = *(const v8h*)(p + 8);
  v16h r;
#pragma unroll
  for (int i = 0; i < 8; ++i) { r[i] = lo[i]; r[i + 8] = hi[i]; }
  return r;
}

// ---------------------------------------------------------------------------
__global__ __launch_bounds__(256)
void cvt_f32_f16_v4(const float* __restrict__ src, _Float16* __restrict__ dst, int n4) {
  int i = blockIdx.x * blockDim.x + threadIdx.x;
  if (i < n4) {
    float4 f = ((const float4*)src)[i];
    _Float16* o = dst + 4 * (size_t)i;
    o[0] = (_Float16)f.x; o[1] = (_Float16)f.y;
    o[2] = (_Float16)f.z; o[3] = (_Float16)f.w;
  }
}

// ---------------------------------------------------------------------------
// QKV projection: Y[m,n] = sum_k X[m,k]*W[n,k].
// Block 256 thr (8 waves), tile 128x128, K-step 32, double-buffered async
// global->LDS staging. Wave (wm,wn) of 4x2 grid owns 32x64 => 8 accums,
// 8 WMMAs per K-step with B-fragment reuse across two A-fragments.
// Epilogue scatters to attention layouts:
//   which=0: Q/8 -> (B,H,T,hs)   which=1: K -> (B,H,T,hs)   which=2: V -> (B,H,hs,T)
// ---------------------------------------------------------------------------
#define BM 128
#define BN 128
#define BK 32
#define LDSP 40   // padded LDS row pitch (halves): 20-dword bank stride, conflict-free

__global__ __launch_bounds__(256)
void qkv_proj_kernel(const _Float16* __restrict__ xh,
                     const _Float16* __restrict__ wh,
                     _Float16* __restrict__ qh,
                     _Float16* __restrict__ kh,
                     _Float16* __restrict__ vt) {
  __shared__ _Float16 sA[2][BM * LDSP];
  __shared__ _Float16 sB[2][BN * LDSP];

  const int tid   = threadIdx.x;
  const int lane  = tid & 31;
  const int wave  = tid >> 5;
  const int wm    = wave >> 1;   // 0..3 : M sub-tile
  const int wn    = wave & 1;    // 0..1 : N sub-tile
  const int mblk  = blockIdx.x * BM;
  const int nblk  = blockIdx.y * BN;
  const int which = blockIdx.z;
  const _Float16* W = wh + (size_t)which * EMBED * EMBED;

  const _Float16* gA0 = xh + (size_t)mblk * EMBED;
  const _Float16* gB0 = W  + (size_t)nblk * EMBED;

  const int mloc = lane & 15;
  const int kA   = (lane < 16) ? 0 : 16;
  const int kB   = (lane < 16) ? 0 : 16;

  // 512 16B chunks per 128x32 tile; thread covers chunks {tid, tid+256}
  auto fill = [&](int bufidx, int k0) {
    _Float16* la = sA[bufidx];
    _Float16* lb = sB[bufidx];
#pragma unroll
    for (int hh = 0; hh < 2; ++hh) {
      const int ch  = tid + hh * 256;
      const int row = ch >> 2;
      const int c8  = (ch & 3) * 8;
      ACP(la + row * LDSP + c8, gA0 + (size_t)row * EMBED + k0 + c8);
      ACP(lb + row * LDSP + c8, gB0 + (size_t)row * EMBED + k0 + c8);
    }
  };

  v8f zero = {0.f,0.f,0.f,0.f,0.f,0.f,0.f,0.f};
  v8f acc[2][4];
#pragma unroll
  for (int mi = 0; mi < 2; ++mi)
#pragma unroll
    for (int nt = 0; nt < 4; ++nt) acc[mi][nt] = zero;

  fill(0, 0);
  const int NK = EMBED / BK;   // 32
  for (int i = 0; i < NK; ++i) {
    const int buf = i & 1;
    if (i + 1 < NK) { fill(buf ^ 1, (i + 1) * BK); WAIT_ASYNC(4); }
    else            { WAIT_ASYNC(0); }
    __syncthreads();   // prior buffer's fills visible block-wide

    const v16h a0 = make_a_frag(&sA[buf][(wm * 32 + mloc) * LDSP], kA);
    const v16h a1 = make_a_frag(&sA[buf][(wm * 32 + 16 + mloc) * LDSP], kA);
#pragma unroll
    for (int nt = 0; nt < 4; ++nt) {
      const v16h b = ld_b16(&sB[buf][(wn * 64 + nt * 16 + mloc) * LDSP + kB]);
      acc[0][nt] = __builtin_amdgcn_wmma_f32_16x16x32_f16(false, a0, false, b, (short)0, acc[0][nt], false, false);
      acc[1][nt] = __builtin_amdgcn_wmma_f32_16x16x32_f16(false, a1, false, b, (short)0, acc[1][nt], false, false);
    }
    __syncthreads();   // reads of this buffer done before it is refilled
  }

  // C/D layout: lanes 0-15 -> M=r, N=lane; lanes 16-31 -> M=8+r, N=lane-16
  const int rh = (lane < 16) ? 0 : 8;
#pragma unroll
  for (int mi = 0; mi < 2; ++mi) {
#pragma unroll
    for (int nt = 0; nt < 4; ++nt) {
      const int n = nblk + wn * 64 + nt * 16 + mloc;
      const int h = n >> 6, d = n & 63;
#pragma unroll
      for (int r = 0; r < 8; ++r) {
        const int gm = mblk + wm * 32 + mi * 16 + r + rh;
        const int b  = gm >> 11;          // / SEQ
        const int t  = gm & (SEQ - 1);
        const float val = acc[mi][nt][r];
        const size_t bht = (size_t)(b * HEADS + h) * SEQ + t;
        if (which == 0)      qh[(bht << 6) + d] = (_Float16)(val * 0.125f); // 1/sqrt(64)
        else if (which == 1) kh[(bht << 6) + d] = (_Float16)val;
        else                 vt[((size_t)(b * HEADS + h) * HSIZE + d) * SEQ + t] = (_Float16)val;
      }
    }
  }
}

// ---------------------------------------------------------------------------
// Causal flash attention. grid = (T/128, B*H), block = 128 (4 waves).
// Each wave owns 32 query rows (2 row-subtiles); loops 32-key blocks to the
// diagonal. Shared K/V fragments feed both subtiles: 16 WMMA per key-block
// for the same K/V bytes -> 2x arithmetic intensity vs 16-row waves.
// qbase is 32-aligned so exactly one block (kb == qbase/32) crosses the
// causal diagonal. Online softmax via 16-lane shfl + v_exp; P re-enters
// WMMA A-layout through a wave-private LDS tile per subtile.
// ---------------------------------------------------------------------------
__global__ __launch_bounds__(128)
void attn_kernel(const _Float16* __restrict__ qh,
                 const _Float16* __restrict__ kh,
                 const _Float16* __restrict__ vt,
                 float* __restrict__ out) {
  __shared__ _Float16 plds[4][2][16][32];

  const int lane  = threadIdx.x & 31;
  const int wave  = threadIdx.x >> 5;
  const int bh    = blockIdx.y;
  const int b     = bh / HEADS;
  const int h     = bh % HEADS;
  const int qbase = blockIdx.x * 128 + wave * 32;   // 32-aligned

  const _Float16* Q = qh + (size_t)bh * SEQ * HSIZE;
  const _Float16* K = kh + (size_t)bh * SEQ * HSIZE;
  const _Float16* V = vt + (size_t)bh * HSIZE * SEQ;

  const int mloc = lane & 15;
  const int kA   = (lane < 16) ? 0 : 16;
  const int kB   = (lane < 16) ? 0 : 16;
  const int rh   = (lane < 16) ? 0 : 8;

  v16h qa[2][2];
#pragma unroll
  for (int u = 0; u < 2; ++u) {
    const _Float16* qrow = Q + (size_t)(qbase + u * 16 + mloc) * HSIZE;
    qa[u][0] = make_a_frag(qrow, kA);        // head-dim 0..31
    qa[u][1] = make_a_frag(qrow + 32, kA);   // head-dim 32..63
  }

  v8f zero = {0.f,0.f,0.f,0.f,0.f,0.f,0.f,0.f};
  v8f o[2][4];
  float mrun[2][8], lrun[2][8];
#pragma unroll
  for (int u = 0; u < 2; ++u) {
#pragma unroll
    for (int c = 0; c < 4; ++c) o[u][c] = zero;
#pragma unroll
    for (int r = 0; r < 8; ++r) { mrun[u][r] = -1e30f; lrun[u][r] = 0.f; }
  }

  const int kbmax = qbase >> 5;
  for (int kb = 0; kb <= kbmax; ++kb) {
    const int kt = kb * 32;

    if (kb < kbmax) {  // prefetch next K/V block (global_prefetch_b8)
      __builtin_prefetch(K + (size_t)(kt + 32 + mloc) * HSIZE, 0, 1);
      __builtin_prefetch(V + (size_t)mloc * SEQ + kt + 32, 0, 1);
    }

    // ---- shared K fragments for this 32-key block ----
    const _Float16* kr0 = K + (size_t)(kt + mloc) * HSIZE;
    const _Float16* kr1 = K + (size_t)(kt + 16 + mloc) * HSIZE;
    const v16h k00 = *(const v16h*)(kr0 + kB);        // tile0, hd 0..31
    const v16h k01 = *(const v16h*)(kr0 + 32 + kB);   // tile0, hd 32..63
    const v16h k10 = *(const v16h*)(kr1 + kB);        // tile1, hd 0..31
    const v16h k11 = *(const v16h*)(kr1 + 32 + kB);   // tile1, hd 32..63

    // ---- S = Q K^T : 2 subtiles x 2 key-tiles ----
    v8f s[2][2];
#pragma unroll
    for (int u = 0; u < 2; ++u) {
      s[u][0] = zero;
      s[u][0] = __builtin_amdgcn_wmma_f32_16x16x32_f16(false, qa[u][0], false, k00, (short)0, s[u][0], false, false);
      s[u][0] = __builtin_amdgcn_wmma_f32_16x16x32_f16(false, qa[u][1], false, k01, (short)0, s[u][0], false, false);
      s[u][1] = zero;
      s[u][1] = __builtin_amdgcn_wmma_f32_16x16x32_f16(false, qa[u][0], false, k10, (short)0, s[u][1], false, false);
      s[u][1] = __builtin_amdgcn_wmma_f32_16x16x32_f16(false, qa[u][1], false, k11, (short)0, s[u][1], false, false);
    }

    // ---- causal mask: only the diagonal block ----
    if (kb == kbmax) {
#pragma unroll
      for (int u = 0; u < 2; ++u) {
#pragma unroll
        for (int r = 0; r < 8; ++r) {
          const int qg = qbase + u * 16 + r + rh;
          if (kt + mloc > qg)      s[u][0][r] = -1e30f;
          if (kt + 16 + mloc > qg) s[u][1][r] = -1e30f;
        }
      }
    }

    // ---- online softmax per subtile ----
    float alpha[2][8];
#pragma unroll
    for (int u = 0; u < 2; ++u) {
#pragma unroll
      for (int r = 0; r < 8; ++r) {
        float mx = fmaxf(s[u][0][r], s[u][1][r]);
        mx = fmaxf(mx, __shfl_xor(mx, 1, 16));
        mx = fmaxf(mx, __shfl_xor(mx, 2, 16));
        mx = fmaxf(mx, __shfl_xor(mx, 4, 16));
        mx = fmaxf(mx, __shfl_xor(mx, 8, 16));
        const float mnew = fmaxf(mrun[u][r], mx);
        alpha[u][r] = __expf(mrun[u][r] - mnew);
        mrun[u][r]  = mnew;
        const float p0 = __expf(s[u][0][r] - mnew);
        const float p1 = __expf(s[u][1][r] - mnew);
        s[u][0][r] = p0; s[u][1][r] = p1;
        float rs = p0 + p1;
        rs += __shfl_xor(rs, 1, 16);
        rs += __shfl_xor(rs, 2, 16);
        rs += __shfl_xor(rs, 4, 16);
        rs += __shfl_xor(rs, 8, 16);
        lrun[u][r] = lrun[u][r] * alpha[u][r] + rs;
      }
    }

    // ---- P (C-layout f32) -> LDS f16 -> A-layout fragments ----
#pragma unroll
    for (int u = 0; u < 2; ++u) {
#pragma unroll
      for (int r = 0; r < 8; ++r) {
        plds[wave][u][r + rh][mloc]      = (_Float16)s[u][0][r];
        plds[wave][u][r + rh][16 + mloc] = (_Float16)s[u][1][r];
      }
    }
    v16h pa[2];
#pragma unroll
    for (int u = 0; u < 2; ++u) pa[u] = make_a_frag(&plds[wave][u][mloc][0], kA);

    // ---- rescale O; shared V fragments; O += P V ----
#pragma unroll
    for (int u = 0; u < 2; ++u)
#pragma unroll
      for (int c = 0; c < 4; ++c)
#pragma unroll
        for (int r = 0; r < 8; ++r) o[u][c][r] *= alpha[u][r];

    v16h vb[4];
#pragma unroll
    for (int c = 0; c < 4; ++c)
      vb[c] = *(const v16h*)(V + (size_t)(c * 16 + mloc) * SEQ + kt + kB);
#pragma unroll
    for (int u = 0; u < 2; ++u)
#pragma unroll
      for (int c = 0; c < 4; ++c)
        o[u][c] = __builtin_amdgcn_wmma_f32_16x16x32_f16(false, pa[u], false, vb[c], (short)0, o[u][c], false, false);
  }

  // ---- normalize and write out[b][t][h*64+d] (fp32) ----
#pragma unroll
  for (int u = 0; u < 2; ++u) {
#pragma unroll
    for (int r = 0; r < 8; ++r) {
      const float inv = 1.0f / lrun[u][r];
      const int qg = qbase + u * 16 + r + rh;
      float* orow = out + ((size_t)b * SEQ + qg) * EMBED + h * 64;
      orow[0 * 16 + mloc] = o[u][0][r] * inv;
      orow[1 * 16 + mloc] = o[u][1][r] * inv;
      orow[2 * 16 + mloc] = o[u][2][r] * inv;
      orow[3 * 16 + mloc] = o[u][3][r] * inv;
    }
  }
}

// ---------------------------------------------------------------------------
extern "C" void kernel_launch(void* const* d_in, const int* in_sizes, int n_in,
                              void* d_out, int out_size, void* d_ws, size_t ws_size,
                              hipStream_t stream) {
  const float* x  = (const float*)d_in[0];
  const float* Wq = (const float*)d_in[1];
  const float* Wk = (const float*)d_in[2];
  const float* Wv = (const float*)d_in[3];
  float* out = (float*)d_out;

  const size_t NX = (size_t)BATCH * SEQ * EMBED;  // 8,388,608
  const size_t NW = (size_t)EMBED * EMBED;        // 1,048,576

  _Float16* ws = (_Float16*)d_ws;
  _Float16* xh = ws;                 // NX halves
  _Float16* wh = xh + NX;            // 3*NW halves (Wq|Wk|Wv)
  _Float16* qh = wh + 3 * NW;        // NX halves, (B,H,T,hs)
  _Float16* kh = qh + NX;            // NX halves, (B,H,T,hs)
  _Float16* vt = kh + NX;            // NX halves, (B,H,hs,T)

  cvt_f32_f16_v4<<<(int)((NX / 4 + 255) / 256), 256, 0, stream>>>(x, xh, (int)(NX / 4));
  cvt_f32_f16_v4<<<(int)((NW / 4 + 255) / 256), 256, 0, stream>>>(Wq, wh, (int)(NW / 4));
  cvt_f32_f16_v4<<<(int)((NW / 4 + 255) / 256), 256, 0, stream>>>(Wk, wh + NW, (int)(NW / 4));
  cvt_f32_f16_v4<<<(int)((NW / 4 + 255) / 256), 256, 0, stream>>>(Wv, wh + 2 * NW, (int)(NW / 4));

  qkv_proj_kernel<<<dim3((BATCH * SEQ) / BM, EMBED / BN, 3), 256, 0, stream>>>(
      xh, wh, qh, kh, vt);

  attn_kernel<<<dim3(SEQ / 128, BATCH * HEADS), 128, 0, stream>>>(qh, kh, vt, out);
}
